// HandProcess_86543591014826
// MI455X (gfx1250) — matrix operand order repaired and checked by reading, not compiled, not run on hardware
//
#include <hip/hip_runtime.h>

// ---- problem constants ----
#define KX      3720      // obs dim in x
#define NOUT    256       // hidden
#define MTILE   64        // rows per workgroup
#define RAWS    260       // raw LDS row stride in dwords (256 floats + 4 pad)
#define ABFS    264       // A bf16 LDS row stride in halves (256 + 8 pad) = 528B
#define NSC     16        // super-chunks (15 x 224K + 1 x 256K tail) => K' = 3616
#define KPADCH  113       // total 32-wide K chunks

typedef __attribute__((ext_vector_type(16))) __bf16        v16bf;
typedef __attribute__((ext_vector_type(8)))  float          v8f;
typedef __attribute__((ext_vector_type(4)))  unsigned int   u32x4;
typedef __attribute__((ext_vector_type(2)))  unsigned int   u32x2;
typedef __attribute__((ext_vector_type(4)))  float          f32x4;
typedef __attribute__((ext_vector_type(8)))  unsigned short u16x8;

union FragAB { v16bf v; u32x4 q[2]; };

static __device__ __forceinline__ unsigned short f2bf_rne(float f) {
    unsigned u = __builtin_bit_cast(unsigned, f);
    u += 0x7FFFu + ((u >> 16) & 1u);
    return (unsigned short)(u >> 16);
}

// ---------------------------------------------------------------------------
// Kernel 1: W (fp32, 256 x 3720) -> bf16 fragment-ordered, K-permuted buffer.
// K' map: k' < 3556 -> k = k'+36 (agents); 3556<=k'<3592 -> k = k'-3556 (self);
//         k' >= 3592 -> zero pad.  Layout: [chunk][ntile][lane][16 halves],
// lane's halves h=0..15 hold K' = chunk*32 + 16*(lane/16) + h for col
// n = ntile*16 + lane%16  (contiguous k order, same as the A-tile reads).
// ---------------------------------------------------------------------------
__global__ __launch_bounds__(256) void conv_w_kernel(
    const float* __restrict__ W, unsigned short* __restrict__ wfrag) {
    int t = blockIdx.x * 256 + threadIdx.x;            // 113*16*32*2 = 115712
    if (t >= KPADCH * 16 * 32 * 2) return;
    int h0    = (t & 1) * 8;
    int lane  = (t >> 1) & 31;
    int tile  = t >> 6;                                 // chunk*16 + ntile
    int chunk = tile >> 4, nt = tile & 15;
    int n     = nt * 16 + (lane & 15);
    int kp0   = chunk * 32 + (lane >> 4) * 16 + h0;
    u16x8 hs;
#pragma unroll
    for (int j = 0; j < 8; ++j) {
        int kp = kp0 + j;
        float val = 0.f;
        int k = (kp < 3556) ? (kp + 36) : ((kp < 3592) ? (kp - 3556) : -1);
        if (k >= 0) val = W[(size_t)n * KX + k];
        hs[j] = f2bf_rne(val);
    }
    *(u16x8*)(wfrag + (size_t)t * 8) = hs;              // 16B aligned store
}

// ---------------------------------------------------------------------------
// Kernel 2: fused hand-attention + GEMM.  512 WGs x 256 threads (8 waves).
// ---------------------------------------------------------------------------
__global__ __launch_bounds__(256) void fused_attn_gemm_kernel(
    const float* __restrict__ x, const unsigned short* __restrict__ wfrag,
    const float* __restrict__ bias, float* __restrict__ out) {

    __shared__ float          raw[MTILE * RAWS];        // 66560 B
    __shared__ unsigned short abf[MTILE * ABFS];        // 33792 B
    __shared__ float          atts[MTILE * 8];          //  2048 B

    const int tid  = threadIdx.x;
    const int lane = tid & 31, wave = tid >> 5;
    const int g    = lane >> 4, nrow = lane & 15;
    const int mbase = blockIdx.x * MTILE;

    v8f acc[4][2];
#pragma unroll
    for (int i = 0; i < 4; ++i)
#pragma unroll
        for (int j = 0; j < 2; ++j)
            acc[i][j] = (v8f){0.f,0.f,0.f,0.f,0.f,0.f,0.f,0.f};

#pragma unroll 1
    for (int sc = 0; sc < NSC; ++sc) {
        const bool tail = (sc == NSC - 1);

        // ---- stage x tile (fp32) into LDS, read-once ----
        if (!tail) {
            const float* src = x + (size_t)mbase * KX + 36 + sc * 224;
#pragma unroll
            for (int j = 0; j < 14; ++j) {              // 64 rows * 56 float4
                int idx = j * 256 + tid;
                int r = idx / 56, c4 = idx % 56;
                f32x4 v = *(const f32x4*)(src + (size_t)r * KX + c4 * 4);
                *(f32x4*)(raw + r * RAWS + c4 * 4) = v;
            }
        } else {
            const float* xr = x + (size_t)mbase * KX;
#pragma unroll
            for (int j = 0; j < 16; ++j) {              // 64 rows * 64 float4
                int idx = j * 256 + tid;
                int r = idx >> 6, c4 = idx & 63;
                f32x4 v = (f32x4){0.f, 0.f, 0.f, 0.f};
                if (c4 < 49)       v = *(const f32x4*)(xr + (size_t)r * KX + 3396 + c4 * 4);
                else if (c4 < 58)  v = *(const f32x4*)(xr + (size_t)r * KX + (c4 - 49) * 4);
                *(f32x4*)(raw + r * RAWS + c4 * 4) = v;
            }
        }
        __syncthreads();

        // ---- hand-attention weights: att = 2^-( |6-argmax_x| + |6-argmax_y| ) ----
        {
            const int nag = tail ? 7 : 8;
            for (int p = tid; p < nag * 64; p += 256) {
                int a = p >> 6, r = p & 63;
                const float* bptr = raw + r * RAWS + a * 28;
                int bi = 0, bj = 0;
                float bx = bptr[2], by = bptr[15];
#pragma unroll
                for (int j = 1; j < 13; ++j) {
                    float vx = bptr[2 + j];  if (vx > bx) { bx = vx; bi = j; }
                    float vy = bptr[15 + j]; if (vy > by) { by = vy; bj = j; }
                }
                int dx = bi > 6 ? bi - 6 : 6 - bi;
                int dy = bj > 6 ? bj - 6 : 6 - bj;
                atts[r * 8 + a] =
                    __builtin_bit_cast(float, (unsigned)(127 - dx - dy) << 23);
            }
        }
        __syncthreads();

        // ---- scale + convert to bf16 A-tile ----
        {
            const int r = tid >> 2, q = tid & 3;
            const int ncols = tail ? 64 : 56;
            const int cbase = q * ncols;
            const float* rp = raw + r * RAWS + cbase;
            unsigned short* op = abf + r * ABFS + cbase;
            for (int c4 = 0; c4 < ncols; c4 += 4) {
                f32x4 v = *(const f32x4*)(rp + c4);
                unsigned short hs[4];
#pragma unroll
                for (int j = 0; j < 4; ++j) {
                    int cc = cbase + c4 + j;
                    float s = (!tail) ? atts[r * 8 + cc / 28]
                              : ((cc < 196) ? atts[r * 8 + cc / 28] : 1.0f);
                    hs[j] = f2bf_rne(v[j] * s);
                }
                u32x2 pk;
                pk.x = (unsigned)hs[0] | ((unsigned)hs[1] << 16);
                pk.y = (unsigned)hs[2] | ((unsigned)hs[3] << 16);
                *(u32x2*)(op + c4) = pk;
            }
        }
        __syncthreads();

        // ---- WMMA over this super-chunk ----
        const int nch = tail ? 8 : 7;
        for (int kc = 0; kc < nch; ++kc) {
            FragAB A[4];
#pragma unroll
            for (int ms = 0; ms < 4; ++ms) {            // A: halves0-15=K 16g..16g+15
                const unsigned short* ap = abf + (ms * 16 + nrow) * ABFS + kc * 32;
                A[ms].q[0] = *(const u32x4*)(ap + 16 * g);
                A[ms].q[1] = *(const u32x4*)(ap + 16 * g + 8);
            }
            FragAB B[2];
            const int kcg = sc * 7 + kc;
#pragma unroll
            for (int nt = 0; nt < 2; ++nt) {            // B: contiguous 32B per lane
                const unsigned short* bp =
                    wfrag + (((size_t)(kcg * 16 + (wave * 2 + nt))) * 32 + lane) * 16;
                B[nt].q[0] = *(const u32x4*)(bp);
                B[nt].q[1] = *(const u32x4*)(bp + 8);
            }
#pragma unroll
            for (int ms = 0; ms < 4; ++ms)
#pragma unroll
                for (int nt = 0; nt < 2; ++nt)
                    acc[ms][nt] = __builtin_amdgcn_wmma_f32_16x16x32_bf16(
                        false, A[ms].v, false, B[nt].v,
                        (short)0, acc[ms][nt], false, false);
        }
        __syncthreads();   // fence abf/raw reuse vs next super-chunk writes
    }

    // ---- epilogue: bias + store (C layout: lanes0-15 M=v, lanes16-31 M=8+v) ----
#pragma unroll
    for (int nt = 0; nt < 2; ++nt) {
        const int nc = (wave * 2 + nt) * 16 + nrow;
        const float bv = bias[nc];
#pragma unroll
        for (int ms = 0; ms < 4; ++ms) {
            const int r0 = mbase + ms * 16 + 8 * g;
#pragma unroll
            for (int v = 0; v < 8; ++v)
                out[(size_t)(r0 + v) * NOUT + nc] = acc[ms][nt][v] + bv;
        }
    }
}

extern "C" void kernel_launch(void* const* d_in, const int* in_sizes, int n_in,
                              void* d_out, int out_size, void* d_ws, size_t ws_size,
                              hipStream_t stream) {
    const float* x = (const float*)d_in[0];
    const float* W = (const float*)d_in[1];
    const float* b = (const float*)d_in[2];
    float* out = (float*)d_out;
    unsigned short* wfrag = (unsigned short*)d_ws;      // 113*16*32*16 halves ≈ 1.85 MB

    hipLaunchKernelGGL(conv_w_kernel, dim3(452), dim3(256), 0, stream, W, wfrag);
    hipLaunchKernelGGL(fused_attn_gemm_kernel, dim3(32768 / MTILE), dim3(256), 0, stream,
                       x, wfrag, b, out);
}
